// Relation_68728066670625
// MI455X (gfx1250) — matrix operand order repaired and checked by reading, not compiled, not run on hardware
//
#include <hip/hip_runtime.h>

#define NQ   2048   // queries
#define MKV  1024   // keys / values
#define DDIM 1024   // feature dim
#define EDIM 64     // pos-embed dim
#define GRP  16     // groups (head dim = 64)

typedef _Float16 h16;
typedef __attribute__((ext_vector_type(16))) _Float16 v16h;
typedef __attribute__((ext_vector_type(8)))  _Float16 v8h;
typedef __attribute__((ext_vector_type(8)))  float    v8f;

#define DEV static __device__ __forceinline__

DEV v8f wmma_f16(v16h a, v16h b, v8f c) {
  return __builtin_amdgcn_wmma_f32_16x16x32_f16(false, a, false, b, (short)0, c,
                                                false, false);
}

// ---- fragment builders (CDNA5 16x16x32 f16 WMMA layouts) -------------------
// A (16x32, MxK): lane row = l&15; lanes<16 hold K {0..7,16..23}, lanes>=16 +8.
DEV v16h frag_a_h(const h16* __restrict__ base, int stride) {
  int l = threadIdx.x & 31;
  const h16* p = base + (size_t)(l & 15) * stride + ((l >> 4) << 3);
  v8h lo = *(const v8h*)p;        // K = koff + 0..7
  v8h hi = *(const v8h*)(p + 16); // K = koff + 16..23
  v16h r;
#pragma unroll
  for (int i = 0; i < 8; ++i) { r[i] = lo[i]; r[i + 8] = hi[i]; }
  return r;
}

DEV v16h frag_a_f32(const float* __restrict__ base, int stride) {
  int l = threadIdx.x & 31;
  const float* p = base + (size_t)(l & 15) * stride + ((l >> 4) << 3);
  v16h r;
#pragma unroll
  for (int i = 0; i < 8; ++i) { r[i] = (h16)p[i]; r[i + 8] = (h16)p[16 + i]; }
  return r;
}

// B (32x16, KxN) with B[k][n] = base[n*stride + k] (K contiguous in memory):
// lane col = l&15; lanes<16 hold K {0..7,16..23}, lanes>=16 +8.
DEV v16h frag_b_h(const h16* __restrict__ base, int stride) {
  int l = threadIdx.x & 31;
  const h16* p = base + (size_t)(l & 15) * stride + ((l >> 4) << 3);
  v8h lo = *(const v8h*)p;
  v8h hi = *(const v8h*)(p + 16);
  v16h r;
#pragma unroll
  for (int i = 0; i < 8; ++i) { r[i] = lo[i]; r[i + 8] = hi[i]; }
  return r;
}

// B with B[k][n] = base[k*stride + n] (K strided), f32 source, cvt to f16.
DEV v16h frag_b_strided_f32(const float* __restrict__ base, size_t stride) {
  int l = threadIdx.x & 31;
  const float* p = base + (l & 15) + (size_t)((l >> 4) << 3) * stride;
  v16h r;
#pragma unroll
  for (int i = 0; i < 16; ++i) r[i] = (h16)p[(size_t)(i + ((i >> 3) << 3)) * stride];
  return r;
}

// ---- kernel 1: cast the four DxD weight matrices to f16 --------------------
__global__ void cvt_kernel(const float* __restrict__ fc, const float* __restrict__ wq,
                           const float* __restrict__ wk, const float* __restrict__ wvw,
                           h16* __restrict__ fch, h16* __restrict__ wqh,
                           h16* __restrict__ wkh, h16* __restrict__ wvh) {
  int i = blockIdx.x * blockDim.x + threadIdx.x;
  if (i < DDIM * DDIM) {
    fch[i] = (h16)fc[i]; wqh[i] = (h16)wq[i];
    wkh[i] = (h16)wk[i]; wvh[i] = (h16)wvw[i];
  }
}

// ---- kernel 2: shared fc + relu on [ref; sup] ------------------------------
// act = relu(X @ fc_w^T + fc_b) stored f16; ref rows also stored f32 (residual)
__global__ __launch_bounds__(128) void fc_kernel(
    const float* __restrict__ ref_feat, const float* __restrict__ sup_feat,
    const h16* __restrict__ fcw, const float* __restrict__ fcb,
    float* __restrict__ ref32, h16* __restrict__ act) {
  int l = threadIdx.x & 31;
  int wid = blockIdx.x * 4 + (threadIdx.x >> 5);
  int tr = wid >> 6;           // 192 row tiles (ref: 0..127, sup: 128..191)
  int tc = wid & 63;           // 64 col tiles
  if (tr >= (NQ + MKV) / 16) return;
  bool is_ref = (tr * 16) < NQ;
  const float* Ab = is_ref ? ref_feat + (size_t)(tr * 16) * DDIM
                           : sup_feat + (size_t)(tr * 16 - NQ) * DDIM;
  const h16* Wb = fcw + (size_t)(tc * 16) * DDIM;
  v8f acc = {0.f, 0.f, 0.f, 0.f, 0.f, 0.f, 0.f, 0.f};
  for (int k0 = 0; k0 < DDIM; k0 += 32)
    acc = wmma_f16(frag_a_f32(Ab + k0, DDIM), frag_b_h(Wb + k0, DDIM), acc);
  int ln = l & 15, hf = l >> 4;
  int jc = tc * 16 + ln;
  float bv = fcb[jc];
#pragma unroll
  for (int r = 0; r < 8; ++r) {
    size_t row = tr * 16 + r + hf * 8;
    float v = fmaxf(acc[r] + bv, 0.f);
    act[row * DDIM + jc] = (h16)v;
    if (is_ref) ref32[row * DDIM + jc] = v;
  }
}

// ---- kernel 3: generic GEMM out = (A @ W^T + b) * scale, f16 out -----------
__global__ __launch_bounds__(128) void gemm_h(
    const h16* __restrict__ A, const h16* __restrict__ W,
    const float* __restrict__ bias, float scale, h16* __restrict__ outp,
    int rows, int trans) {
  int l = threadIdx.x & 31;
  int wid = blockIdx.x * 4 + (threadIdx.x >> 5);
  int tcnt = DDIM / 16;
  int tr = wid / tcnt, tc = wid % tcnt;
  if (tr >= rows / 16) return;
  const h16* Ab = A + (size_t)(tr * 16) * DDIM;
  const h16* Wb = W + (size_t)(tc * 16) * DDIM;
  v8f acc = {0.f, 0.f, 0.f, 0.f, 0.f, 0.f, 0.f, 0.f};
  for (int k0 = 0; k0 < DDIM; k0 += 32)
    acc = wmma_f16(frag_a_h(Ab + k0, DDIM), frag_b_h(Wb + k0, DDIM), acc);
  int ln = l & 15, hf = l >> 4;
  int jc = tc * 16 + ln;
  float bv = bias ? bias[jc] : 0.f;
#pragma unroll
  for (int r = 0; r < 8; ++r) {
    size_t row = tr * 16 + r + hf * 8;
    float v = (acc[r] + bv) * scale;
    if (trans) outp[(size_t)jc * rows + row] = (h16)v;   // V^T: [D][M]
    else       outp[row * DDIM + jc] = (h16)v;
  }
}

// ---- kernel 4: position bias: L1[n,g,m] = log(relu(wg_w@pe + wg_b) + 1e-6) -
// Per n: [G=16,E=64] @ [E, m-tile] WMMA; streams the 537 MB pos_emb once.
__global__ __launch_bounds__(128) void posbias_kernel(
    const float* __restrict__ wg_w, const float* __restrict__ wg_b,
    const float* __restrict__ pe, float* __restrict__ L1) {
  int l = threadIdx.x & 31;
  int wid = blockIdx.x * 4 + (threadIdx.x >> 5);
  int n = wid >> 6, mt = wid & 63;
  if (n >= NQ) return;
  const size_t NM = (size_t)NQ * MKV;
  const float* pb = pe + (size_t)n * MKV + mt * 16;  // pe[e][n][m0+..]
  v8f acc = {0.f, 0.f, 0.f, 0.f, 0.f, 0.f, 0.f, 0.f};
  acc = wmma_f16(frag_a_f32(wg_w, EDIM),      frag_b_strided_f32(pb, NM), acc);
  acc = wmma_f16(frag_a_f32(wg_w + 32, EDIM), frag_b_strided_f32(pb + 32 * NM, NM), acc);
  int ln = l & 15, hf = l >> 4;
#pragma unroll
  for (int r = 0; r < 8; ++r) {
    int g = r + hf * 8;
    float x = fmaxf(acc[r] + wg_b[g], 0.f);
    L1[((size_t)n * GRP + g) * MKV + mt * 16 + ln] = __logf(x + 1e-6f);
  }
}

// ---- kernel 5: fused flash attention -------------------------------------
// wave = (16-row n-tile, group g): logits = L1 + QK^T (L1 fused as WMMA C),
// online softmax, P re-layout via LDS, O += P @ V_g. out = ref + O/sum + wv_b
__global__ __launch_bounds__(128) void attn_kernel(
    const h16* __restrict__ q_h, const h16* __restrict__ k_h,
    const h16* __restrict__ v_t, const float* __restrict__ L1,
    const float* __restrict__ ref32, const float* __restrict__ wvb,
    float* __restrict__ outp) {
  __shared__ h16 Pst[4][16 * 32];
  int l = threadIdx.x & 31;
  int wvid = threadIdx.x >> 5;
  int wid = blockIdx.x * 4 + wvid;
  int nt = wid >> 4;   // 0..127
  int g  = wid & 15;
  if (nt >= NQ / 16) return;
  int ln = l & 15, hf = l >> 4;

  const h16* qb = q_h + (size_t)(nt * 16) * DDIM + g * 64;
  v16h qa0 = frag_a_h(qb, DDIM);
  v16h qa1 = frag_a_h(qb + 32, DDIM);

  v8f o0 = {0.f,0.f,0.f,0.f,0.f,0.f,0.f,0.f};
  v8f o1 = o0, o2 = o0, o3 = o0;
  float rmax[8], rsum[8];
#pragma unroll
  for (int r = 0; r < 8; ++r) { rmax[r] = -3.0e38f; rsum[r] = 0.f; }

  for (int c = 0; c < MKV / 32; ++c) {
    int mb = c * 32;
    // seed logits with log-position-bias (C operand of QK^T WMMA)
    v8f s0, s1;
#pragma unroll
    for (int r = 0; r < 8; ++r) {
      size_t nrow = (size_t)(nt * 16 + r + hf * 8);
      const float* lp = L1 + (nrow * GRP + g) * MKV + mb + ln;
      s0[r] = lp[0];
      s1[r] = lp[16];
    }
    const h16* kb = k_h + (size_t)mb * DDIM + g * 64;
    s0 = wmma_f16(qa0, frag_b_h(kb, DDIM), s0);
    s0 = wmma_f16(qa1, frag_b_h(kb + 32, DDIM), s0);
    const h16* kb2 = kb + (size_t)16 * DDIM;
    s1 = wmma_f16(qa0, frag_b_h(kb2, DDIM), s1);
    s1 = wmma_f16(qa1, frag_b_h(kb2 + 32, DDIM), s1);

    // online softmax: row stats live in lanes of the same 16-lane half
#pragma unroll
    for (int r = 0; r < 8; ++r) {
      float m2 = fmaxf(s0[r], s1[r]);
      m2 = fmaxf(m2, __shfl_xor(m2, 1));
      m2 = fmaxf(m2, __shfl_xor(m2, 2));
      m2 = fmaxf(m2, __shfl_xor(m2, 4));
      m2 = fmaxf(m2, __shfl_xor(m2, 8));
      float nm = fmaxf(rmax[r], m2);
      float sc = __expf(rmax[r] - nm);
      rmax[r] = nm;
      float p0 = __expf(s0[r] - nm);
      float p1 = __expf(s1[r] - nm);
      rsum[r] = rsum[r] * sc + p0 + p1;
      o0[r] *= sc; o1[r] *= sc; o2[r] *= sc; o3[r] *= sc;
      int rr = r + hf * 8;
      Pst[wvid][rr * 32 + ln]      = (h16)p0;   // C-layout -> LDS
      Pst[wvid][rr * 32 + 16 + ln] = (h16)p1;
    }
    // reload P as an A fragment (16x32)
    v16h pa;
    {
      const h16* pp = &Pst[wvid][ln * 32 + hf * 8];
#pragma unroll
      for (int i = 0; i < 8; ++i) { pa[i] = pp[i]; pa[i + 8] = pp[16 + i]; }
    }
    const h16* vb = v_t + (size_t)(g * 64) * MKV + mb;  // V^T [D][M]
    o0 = wmma_f16(pa, frag_b_h(vb, MKV), o0);
    o1 = wmma_f16(pa, frag_b_h(vb + (size_t)16 * MKV, MKV), o1);
    o2 = wmma_f16(pa, frag_b_h(vb + (size_t)32 * MKV, MKV), o2);
    o3 = wmma_f16(pa, frag_b_h(vb + (size_t)48 * MKV, MKV), o3);
  }

#pragma unroll
  for (int r = 0; r < 8; ++r) {
    float s = rsum[r];
    s += __shfl_xor(s, 1);
    s += __shfl_xor(s, 2);
    s += __shfl_xor(s, 4);
    s += __shfl_xor(s, 8);
    rsum[r] = 1.0f / s;
  }
#pragma unroll
  for (int jt = 0; jt < 4; ++jt) {
    v8f& o = (jt == 0 ? o0 : jt == 1 ? o1 : jt == 2 ? o2 : o3);
    int jc = g * 64 + jt * 16 + ln;
    float bv = wvb[jc];
#pragma unroll
    for (int r = 0; r < 8; ++r) {
      size_t n = (size_t)(nt * 16 + r + hf * 8);
      outp[n * DDIM + jc] = ref32[n * DDIM + jc] + o[r] * rsum[r] + bv;
    }
  }
}

// ---- host-side launch ------------------------------------------------------
extern "C" void kernel_launch(void* const* d_in, const int* in_sizes, int n_in,
                              void* d_out, int out_size, void* d_ws, size_t ws_size,
                              hipStream_t stream) {
  const float* ref_feat = (const float*)d_in[0];
  const float* sup_feat = (const float*)d_in[1];
  const float* pe       = (const float*)d_in[2];
  const float* fc_w     = (const float*)d_in[3];
  const float* fc_b     = (const float*)d_in[4];
  const float* wg_w     = (const float*)d_in[5];
  const float* wg_b     = (const float*)d_in[6];
  const float* wq_w     = (const float*)d_in[7];
  const float* wq_b     = (const float*)d_in[8];
  const float* wk_w     = (const float*)d_in[9];
  const float* wk_b     = (const float*)d_in[10];
  const float* wv_w     = (const float*)d_in[11];
  const float* wv_b     = (const float*)d_in[12];
  float* out = (float*)d_out;

  char* ws = (char*)d_ws;
  auto alloc = [&](size_t bytes) {
    char* p = ws;
    ws += (bytes + 255) & ~(size_t)255;
    return p;
  };
  h16*   fcw_h = (h16*)alloc((size_t)DDIM * DDIM * 2);
  h16*   wqw_h = (h16*)alloc((size_t)DDIM * DDIM * 2);
  h16*   wkw_h = (h16*)alloc((size_t)DDIM * DDIM * 2);
  h16*   wvw_h = (h16*)alloc((size_t)DDIM * DDIM * 2);
  h16*   act_h = (h16*)alloc((size_t)(NQ + MKV) * DDIM * 2);
  float* ref32 = (float*)alloc((size_t)NQ * DDIM * 4);
  h16*   q_h   = (h16*)alloc((size_t)NQ * DDIM * 2);
  h16*   k_h   = (h16*)alloc((size_t)MKV * DDIM * 2);
  h16*   v_t   = (h16*)alloc((size_t)DDIM * MKV * 2);
  float* L1    = (float*)alloc((size_t)NQ * GRP * MKV * 4);

  cvt_kernel<<<(DDIM * DDIM + 255) / 256, 256, 0, stream>>>(
      fc_w, wq_w, wk_w, wv_w, fcw_h, wqw_h, wkw_h, wvw_h);

  // (N+M)/16 * D/16 = 12288 tiles, 4 waves/block
  fc_kernel<<<3072, 128, 0, stream>>>(ref_feat, sup_feat, fcw_h, fc_b, ref32, act_h);

  gemm_h<<<2048, 128, 0, stream>>>(act_h, wqw_h, wq_b, 0.125f, q_h, NQ, 0);   // q/sqrt(64)
  gemm_h<<<1024, 128, 0, stream>>>(act_h + (size_t)NQ * DDIM, wkw_h, wk_b, 1.0f, k_h, MKV, 0);
  gemm_h<<<1024, 128, 0, stream>>>(act_h + (size_t)NQ * DDIM, wvw_h, nullptr, 1.0f, v_t, MKV, 1);

  // N * M/16 = 131072 tiles, 4 waves/block
  posbias_kernel<<<32768, 128, 0, stream>>>(wg_w, wg_b, pe, L1);

  // N/16 * G = 2048 waves, 4 waves/block
  attn_kernel<<<512, 128, 0, stream>>>(q_h, k_h, v_t, L1, ref32, wv_b, out);
}
